// InnerMonologue_23227183137018
// MI455X (gfx1250) — matrix operand order repaired and checked by reading, not compiled, not run on hardware
//
#include <hip/hip_runtime.h>
#include <hip/hip_bf16.h>

// Problem dims (fixed by the reference)
#define H_DIM 2048
#define P_DIM 256
#define B_DIM 4
#define S_DIM 8192
#define M_DIM (B_DIM * S_DIM)   // 32768 rows (b,s flattened)

// Tile config: 256 threads = 8 wave32, wave grid 2(M) x 4(N).
#define MT 64
#define NT 256
#define KT 32

typedef __attribute__((ext_vector_type(16))) __bf16 v16bf;
typedef __attribute__((ext_vector_type(8)))  __bf16 v8bf;
typedef __attribute__((ext_vector_type(4)))  __bf16 v4bf;
typedef __attribute__((ext_vector_type(8)))  float  v8f;

// XOR swizzle: 16B chunk index permuted by row so that the 16 lanes of a
// fragment read hit 16 distinct LDS bank groups (row stride is 64B = 16 banks).
__device__ __forceinline__ int swz(int chunk, int row) {
    return chunk ^ (row & 3) ^ ((row >> 2) & 3);
}

// Assemble a v16bf WMMA fragment from two swizzled 16B LDS chunks of one row.
__device__ __forceinline__ v16bf load_frag(const __bf16* base, int row, int c0, int c1) {
    const v8bf* p0 = (const v8bf*)(base + row * KT + swz(c0, row) * 8);
    const v8bf* p1 = (const v8bf*)(base + row * KT + swz(c1, row) * 8);
    v8bf a = *p0;
    v8bf b = *p1;
    return __builtin_shufflevector(a, b, 0,1,2,3,4,5,6,7,8,9,10,11,12,13,14,15);
}

// Stage one (A 64x32, B 32x256) fp32 k-tile into registers.
__device__ __forceinline__ void load_tile_G(const float* __restrict__ A, int lda, int row0, int kb,
                                            const float* __restrict__ W, int ldw, int nb,
                                            int tid, float4* ra, float4* rb) {
#pragma unroll
    for (int i = 0; i < 2; ++i) {                 // A: 512 float4, 2/thread
        int f  = tid + 256 * i;
        int r  = f >> 3;
        int c4 = f & 7;
        ra[i] = *(const float4*)(A + (size_t)(row0 + r) * lda + kb + c4 * 4);
    }
#pragma unroll
    for (int i = 0; i < 2; ++i) {                 // B: two 4x4 micro-tiles/thread
        int mt = tid + 256 * i;
        int kg = mt >> 6;
        int ng = mt & 63;
#pragma unroll
        for (int r = 0; r < 4; ++r)
            rb[i * 4 + r] = *(const float4*)(W + (size_t)(kb + kg * 4 + r) * ldw + nb + ng * 4);
    }
}

// Convert staged fp32 regs to bf16 and store into swizzled LDS tiles.
// sA: [MT][KT] row-major. sB: [NT][KT] (i.e. W-tile transposed, K contiguous).
__device__ __forceinline__ void store_tile_L(__bf16* sA, __bf16* sB, int tid,
                                             const float4* ra, const float4* rb) {
#pragma unroll
    for (int i = 0; i < 2; ++i) {
        int f  = tid + 256 * i;
        int r  = f >> 3;
        int k  = (f & 7) * 4;
        v4bf v;
        v[0] = (__bf16)ra[i].x; v[1] = (__bf16)ra[i].y;
        v[2] = (__bf16)ra[i].z; v[3] = (__bf16)ra[i].w;
        *(v4bf*)(sA + r * KT + swz(k >> 3, r) * 8 + (k & 7)) = v;
    }
#pragma unroll
    for (int i = 0; i < 2; ++i) {
        int mt = tid + 256 * i;
        int kg = mt >> 6;
        int ng = mt & 63;
        int k  = kg * 4;
#pragma unroll
        for (int d = 0; d < 4; ++d) {             // transpose 4x4 micro-tile in regs
            int n = ng * 4 + d;
            v4bf v;
            v[0] = (__bf16)(((const float*)&rb[i * 4 + 0])[d]);
            v[1] = (__bf16)(((const float*)&rb[i * 4 + 1])[d]);
            v[2] = (__bf16)(((const float*)&rb[i * 4 + 2])[d]);
            v[3] = (__bf16)(((const float*)&rb[i * 4 + 3])[d]);
            *(v4bf*)(sB + n * KT + swz(k >> 3, n) * 8 + (k & 7)) = v;
        }
    }
}

// ---------------------------------------------------------------------------
// Mask kernel: parity prefix over S=8192 (row 0 of token_ids), one block.
// ---------------------------------------------------------------------------
__global__ void __launch_bounds__(256)
mask_kernel(const int* __restrict__ tok, const int* __restrict__ thought_ptr,
            int* __restrict__ mask, float* __restrict__ out_is_private,
            float* __restrict__ out_usage) {
    __shared__ int sc[256];
    const int t = threadIdx.x;
    const int thought = thought_ptr[0];
    const int base = t * 32;

    unsigned bits = 0;
    int c = 0;
#pragma unroll
    for (int i = 0; i < 32; ++i) {
        int m = (tok[base + i] == thought) ? 1 : 0;
        bits |= (unsigned)m << i;
        c += m;
    }
    sc[t] = c;
    __syncthreads();
    for (int off = 1; off < 256; off <<= 1) {     // Hillis-Steele inclusive scan
        int v = 0;
        if (t >= off) v = sc[t - off];
        __syncthreads();
        sc[t] += v;
        __syncthreads();
    }
    int run = sc[t] - c;                          // exclusive prefix count
    int total = sc[255];
#pragma unroll
    for (int i = 0; i < 32; ++i) {
        run += (bits >> i) & 1;                   // toggle BEFORE use
        int par = run & 1;
        mask[base + i] = par;
        float pf = (float)par;
#pragma unroll
        for (int b = 0; b < B_DIM; ++b)
            out_is_private[(size_t)b * S_DIM + base + i] = pf;
    }
    if (t < B_DIM) out_usage[t] = (float)total / (float)S_DIM;
}

// ---------------------------------------------------------------------------
// GEMM1: PR[M,256] = hidden[M,2048] x W_to[2048,256] + b_to   (bf16 WMMA, f32 acc)
// ---------------------------------------------------------------------------
__global__ void __launch_bounds__(256)
gemm1_kernel(const float* __restrict__ hidden, const float* __restrict__ Wto,
             const float* __restrict__ bto, float* __restrict__ pr) {
    __shared__ __bf16 sA[2][MT * KT];
    __shared__ __bf16 sB[2][NT * KT];

    const int tid = threadIdx.x;
    const int wid = tid >> 5, lane = tid & 31;
    const int wm = wid >> 2, wn = wid & 3;
    const int h = lane >> 4, l16 = lane & 15;
    const int row0 = blockIdx.x * MT;

    v8f acc[2][4];
#pragma unroll
    for (int i = 0; i < 2; ++i)
#pragma unroll
        for (int j = 0; j < 4; ++j) acc[i][j] = (v8f){};

    float4 ra[2], rb[8];
    load_tile_G(hidden, H_DIM, row0, 0, Wto, P_DIM, 0, tid, ra, rb);
    store_tile_L(sA[0], sB[0], tid, ra, rb);
    __syncthreads();

    const int NKT = H_DIM / KT;                   // 64 k-tiles
    for (int kt = 0; kt < NKT; ++kt) {
        const int cur = kt & 1, nxt = cur ^ 1;
        if (kt + 1 < NKT)
            load_tile_G(hidden, H_DIM, row0, (kt + 1) * KT, Wto, P_DIM, 0, tid, ra, rb);
        if (kt + 2 < NKT)                          // gfx1250 global_prefetch hint
            __builtin_prefetch(hidden + (size_t)(row0 + (tid >> 3)) * H_DIM + (kt + 2) * KT + (tid & 7) * 4, 0, 0);

        v16bf af[2], bf[4];
#pragma unroll
        for (int i = 0; i < 2; ++i)
            af[i] = load_frag(sA[cur], wm * 32 + i * 16 + l16, h, 2 + h);
#pragma unroll
        for (int j = 0; j < 4; ++j)
            bf[j] = load_frag(sB[cur], wn * 64 + j * 16 + l16, h, 2 + h);
#pragma unroll
        for (int i = 0; i < 2; ++i)
#pragma unroll
            for (int j = 0; j < 4; ++j)
                acc[i][j] = __builtin_amdgcn_wmma_f32_16x16x32_bf16(
                    false, af[i], false, bf[j], (short)0, acc[i][j], false, false);

        if (kt + 1 < NKT) store_tile_L(sA[nxt], sB[nxt], tid, ra, rb);
        __syncthreads();
    }

#pragma unroll
    for (int i = 0; i < 2; ++i)
#pragma unroll
        for (int j = 0; j < 4; ++j) {
            const int col = wn * 64 + j * 16 + l16;
            const float bias = bto[col];
            const int rbase = row0 + wm * 32 + i * 16 + h * 8;
#pragma unroll
            for (int v = 0; v < 8; ++v)
                pr[(size_t)(rbase + v) * P_DIM + col] = acc[i][j][v] + bias;
        }
}

// ---------------------------------------------------------------------------
// GEMM2 + fused blend: out = mask ? PR x W_from + b_from : hidden
// Grid: (M/64, H/256). Pure-public tiles skip the WMMA loop entirely.
// ---------------------------------------------------------------------------
__global__ void __launch_bounds__(256)
gemm2_kernel(const float* __restrict__ pr, const float* __restrict__ Wfrom,
             const float* __restrict__ bfrom, const float* __restrict__ hidden,
             const int* __restrict__ mask, float* __restrict__ outp) {
    __shared__ __bf16 sA[2][MT * KT];
    __shared__ __bf16 sB[2][NT * KT];
    __shared__ int s_rowmask[MT];
    __shared__ int s_any;

    const int tid = threadIdx.x;
    const int wid = tid >> 5, lane = tid & 31;
    const int wm = wid >> 2, wn = wid & 3;
    const int h = lane >> 4, l16 = lane & 15;
    const int row0 = blockIdx.x * MT;
    const int nb = blockIdx.y * NT;

    if (tid == 0) s_any = 0;
    __syncthreads();
    if (tid < MT) {
        int m = mask[(row0 + tid) & (S_DIM - 1)];  // row = b*S + s ; mask depends on s only
        s_rowmask[tid] = m;
        if (m) s_any = 1;
    }
    __syncthreads();

    if (!s_any) {                                  // whole tile public: copy hidden
#pragma unroll
        for (int i = 0; i < 16; ++i) {
            int f = tid + 256 * i;
            int r = f >> 6, c4 = f & 63;
            size_t off = (size_t)(row0 + r) * H_DIM + nb + c4 * 4;
            *(float4*)(outp + off) = *(const float4*)(hidden + off);
        }
        return;
    }

    v8f acc[2][4];
#pragma unroll
    for (int i = 0; i < 2; ++i)
#pragma unroll
        for (int j = 0; j < 4; ++j) acc[i][j] = (v8f){};

    float4 ra[2], rb[8];
    load_tile_G(pr, P_DIM, row0, 0, Wfrom, H_DIM, nb, tid, ra, rb);
    store_tile_L(sA[0], sB[0], tid, ra, rb);
    __syncthreads();

    const int NKT = P_DIM / KT;                    // 8 k-tiles
    for (int kt = 0; kt < NKT; ++kt) {
        const int cur = kt & 1, nxt = cur ^ 1;
        if (kt + 1 < NKT)
            load_tile_G(pr, P_DIM, row0, (kt + 1) * KT, Wfrom, H_DIM, nb, tid, ra, rb);

        v16bf af[2], bf[4];
#pragma unroll
        for (int i = 0; i < 2; ++i)
            af[i] = load_frag(sA[cur], wm * 32 + i * 16 + l16, h, 2 + h);
#pragma unroll
        for (int j = 0; j < 4; ++j)
            bf[j] = load_frag(sB[cur], wn * 64 + j * 16 + l16, h, 2 + h);
#pragma unroll
        for (int i = 0; i < 2; ++i)
#pragma unroll
            for (int j = 0; j < 4; ++j)
                acc[i][j] = __builtin_amdgcn_wmma_f32_16x16x32_bf16(
                    false, af[i], false, bf[j], (short)0, acc[i][j], false, false);

        if (kt + 1 < NKT) store_tile_L(sA[nxt], sB[nxt], tid, ra, rb);
        __syncthreads();
    }

#pragma unroll
    for (int i = 0; i < 2; ++i)
#pragma unroll
        for (int j = 0; j < 4; ++j) {
            const int col = nb + wn * 64 + j * 16 + l16;
            const float bias = bfrom[col];
#pragma unroll
            for (int v = 0; v < 8; ++v) {
                const int rl = wm * 32 + i * 16 + h * 8 + v;
                const int row = row0 + rl;
                float o;
                if (s_rowmask[rl])
                    o = acc[i][j][v] + bias;       // private: projected value
                else
                    o = hidden[(size_t)row * H_DIM + col]; // public: passthrough
                outp[(size_t)row * H_DIM + col] = o;
            }
        }
}

// ---------------------------------------------------------------------------
extern "C" void kernel_launch(void* const* d_in, const int* in_sizes, int n_in,
                              void* d_out, int out_size, void* d_ws, size_t ws_size,
                              hipStream_t stream) {
    const float* hidden  = (const float*)d_in[0];
    const float* Wto     = (const float*)d_in[1];
    const float* bto     = (const float*)d_in[2];
    const float* Wfrom   = (const float*)d_in[3];
    const float* bfrom   = (const float*)d_in[4];
    const int*   tok     = (const int*)d_in[5];
    const int*   thought = (const int*)d_in[6];

    float* out = (float*)d_out;
    // d_out layout = outputs concatenated flat in return order:
    float* pr      = out;                                            // (B,S,P)
    float* outp    = out + (size_t)M_DIM * P_DIM;                    // (B,S,H)
    float* is_priv = outp + (size_t)M_DIM * H_DIM;                   // (B,S)
    float* usage   = is_priv + (size_t)M_DIM;                        // (B,)

    int* mask = (int*)d_ws;                                          // S ints

    mask_kernel<<<1, 256, 0, stream>>>(tok, thought, mask, is_priv, usage);
    gemm1_kernel<<<M_DIM / MT, 256, 0, stream>>>(hidden, Wto, bto, pr);
    gemm2_kernel<<<dim3(M_DIM / MT, H_DIM / NT), 256, 0, stream>>>(
        pr, Wfrom, bfrom, hidden, mask, outp);
}
